// HMRGNN_27084063768650
// MI455X (gfx1250) — matrix operand order repaired and checked by reading, not compiled, not run on hardware
//
#include <hip/hip_runtime.h>

typedef __attribute__((ext_vector_type(16))) _Float16 v16h;
typedef __attribute__((ext_vector_type(8)))  _Float16 v8h;
typedef __attribute__((ext_vector_type(8)))  float    v8f;
typedef __attribute__((ext_vector_type(4)))  int      i32x4;

#define HD 64
#define AS_ 136   // A-tile stride in halves (16B-aligned rows, conflict padding)
#define OS_ 68    // out-tile stride in floats

#define GAS __attribute__((address_space(1)))
#define LAS __attribute__((address_space(3)))

#if defined(__gfx1250__) && __has_builtin(__builtin_amdgcn_global_load_async_to_lds_b128) && __has_builtin(__builtin_amdgcn_s_wait_asynccnt)
#define USE_ASYNC_LDS 1
#else
#define USE_ASYNC_LDS 0
#endif

// ---- async (or fallback) staging of pre-packed f16 weight image into LDS ---
// halves must be a multiple of 2048 (256 threads * 8 halves).
__device__ __forceinline__ void stage_weights(const _Float16* __restrict__ gsrc,
                                              _Float16* __restrict__ lds,
                                              int halves, int tid) {
#if USE_ASYNC_LDS
    for (int off = tid * 8; off < halves; off += 256 * 8) {
        __builtin_amdgcn_global_load_async_to_lds_b128(
            (GAS i32x4*)(unsigned long long)(gsrc + off),
            (LAS i32x4*)(unsigned long long)(lds + off), 0, 0);
    }
    __builtin_amdgcn_s_wait_asynccnt(0);
#else
    for (int off = tid * 8; off < halves; off += 256 * 8)
        *(v8h*)(lds + off) = *(const v8h*)(gsrc + off);
#endif
}

// ---- WMMA fragment helpers (wave32, 16x16x32 f16 -> f32) -------------------
// A: 16x32 (MxK) f16 in LDS, row stride AS_ halves. Per lane: two contiguous
// 16B groups -> 2 x ds_load_b128.
__device__ __forceinline__ v16h frag_A(const _Float16* base, int kBase, int lane) {
    const int row = lane & 15, hi = lane >> 4;
    const _Float16* p = base + row * AS_ + kBase + hi * 16;
    v8h lo = *(const v8h*)p;
    v8h hh = *(const v8h*)(p + 8);
    return __builtin_shufflevector(lo, hh, 0,1,2,3,4,5,6,7,8,9,10,11,12,13,14,15);
}

// B: pre-packed fragment order [chunk][nt][lane][16 halves] -> single v16h load.
__device__ __forceinline__ v16h frag_B(const _Float16* pk, int chunk, int nt, int lane) {
    return *(const v16h*)(pk + ((((chunk << 2) + nt) << 5) + lane) * 16);
}

__device__ __forceinline__ v8f wmma_f16(v16h a, v16h b, v8f c) {
    return __builtin_amdgcn_wmma_f32_16x16x32_f16(false, a, false, b, (short)0, c, false, false);
}

// D tile (16x16 f32): lane 0-15 -> col=lane rows 0-7; lane16-31 -> col=lane-16 rows 8-15.
__device__ __forceinline__ void store_D(float* O, int nBase, int lane, v8f d) {
    const int col = lane & 15, mb = (lane >> 4) * 8;
#pragma unroll
    for (int r = 0; r < 8; ++r) O[(mb + r) * OS_ + nBase + col] = d[r];
}

__device__ __forceinline__ void cvt8(v8h& p, float4 a, float4 b) {
    p[0]=(_Float16)a.x; p[1]=(_Float16)a.y; p[2]=(_Float16)a.z; p[3]=(_Float16)a.w;
    p[4]=(_Float16)b.x; p[5]=(_Float16)b.y; p[6]=(_Float16)b.z; p[7]=(_Float16)b.w;
}

// ---------------------------------------------------------------------------
// Prep: pack a [K,64] f32 weight matrix into WMMA-B fragment order (f16).
// packed[p]: h=p&15, lane=(p>>4)&31, nt=(p>>9)&3, chunk=p>>11
//   k = chunk*32 + (lane>>4)*16 + h ; n = nt*16 + (lane&15)
// ---------------------------------------------------------------------------
__global__ void pack_w_kernel(const float* __restrict__ W, _Float16* __restrict__ out, int K) {
    const int total = K * HD;
    for (int p = blockIdx.x * 256 + threadIdx.x; p < total; p += gridDim.x * 256) {
        const int h = p & 15, lane = (p >> 4) & 31, nt = (p >> 9) & 3, chunk = p >> 11;
        const int k = chunk * 32 + (lane >> 4) * 16 + h;
        const int n = nt * 16 + (lane & 15);
        out[p] = (_Float16)W[k * HD + n];
    }
}

// ---------------------------------------------------------------------------
// K1: per-edge embedding. x_cat[128] = [rel_emb[t] | attr], GEMM vs edge_w
// (128x64, pre-packed), +edge_b, l2norm, leaky -> x_edge (f16). Wave=16 edges.
// ---------------------------------------------------------------------------
__global__ __launch_bounds__(256) void edge_embed_kernel(
    const int* __restrict__ et3, const int* __restrict__ etype,
    const float* __restrict__ ew, const float* __restrict__ rel_emb,
    const float* __restrict__ int_emb,
    const float* __restrict__ rbf0_w, const float* __restrict__ rbf0_b,
    const float* __restrict__ rbf1_w, const float* __restrict__ rbf1_b,
    const _Float16* __restrict__ edge_wp, const float* __restrict__ edge_b,
    _Float16* __restrict__ x_edge, int E)
{
    __shared__ __align__(64) _Float16 sWp[128 * HD];                 // packed, 16 KB
    __shared__ __align__(64) unsigned short sT[8][16 * AS_];         // per-wave tile

    const int tid = threadIdx.x, wave = tid >> 5, lane = tid & 31;
    stage_weights(edge_wp, sWp, 128 * HD, tid);

    const int row = lane & 15, hi = lane >> 4;
    const int eBase = blockIdx.x * 128 + wave * 16;
    int e = eBase + row; if (e >= E) e = E - 1;
    const int t = etype[e];
    const int et0 = et3[0], et1 = et3[1], et2 = et3[2];
    _Float16* A = (_Float16*)&sT[wave][0];

    if (hi == 0) {
        int tr = (t >= 0 && t < 4) ? t : 0;
        const float* re = rel_emb + tr * HD;
#pragma unroll
        for (int j0 = 0; j0 < HD; j0 += 8) {
            v8h p; cvt8(p, *(const float4*)(re + j0), *(const float4*)(re + j0 + 4));
            *(v8h*)(A + row * AS_ + j0) = p;
        }
    } else {
        const float w = ew[e];
        const float* cp = nullptr; const float* Wr = nullptr; const float* Br = nullptr;
        if (t == et0)      { int k = (int)w; k = k < 0 ? 0 : (k > 7 ? 7 : k); cp = int_emb + k * HD; }
        else if (t == et1) { Wr = rbf0_w; Br = rbf0_b; }
        else if (t == et2) { Wr = rbf1_w; Br = rbf1_b; }
        else               { int tr = (t >= 0 && t < 4) ? t : 0; cp = rel_emb + tr * HD; }
        if (cp) {
#pragma unroll
            for (int j0 = 0; j0 < HD; j0 += 8) {
                v8h p; cvt8(p, *(const float4*)(cp + j0), *(const float4*)(cp + j0 + 4));
                *(v8h*)(A + row * AS_ + HD + j0) = p;
            }
        } else {
            float rbf[20];
#pragma unroll
            for (int k = 0; k < 20; ++k) { float d = w - 0.1f * (float)k; rbf[k] = __expf(-10.0f * d * d); }
            for (int j0 = 0; j0 < HD; j0 += 8) {
                v8h p;
#pragma unroll
                for (int jj = 0; jj < 8; ++jj) {
                    float s = Br[j0 + jj];
#pragma unroll
                    for (int k = 0; k < 20; ++k) s = fmaf(rbf[k], Wr[k * HD + j0 + jj], s);
                    p[jj] = (_Float16)s;
                }
                *(v8h*)(A + row * AS_ + HD + j0) = p;
            }
        }
    }
    __syncthreads();

    v16h a0 = frag_A(A, 0, lane),  a1 = frag_A(A, 32, lane),
         a2 = frag_A(A, 64, lane), a3 = frag_A(A, 96, lane);
    v8f d[4];
#pragma unroll
    for (int nt = 0; nt < 4; ++nt) {
        v8f c = {};
        c = wmma_f16(a0, frag_B(sWp, 0, nt, lane), c);
        c = wmma_f16(a1, frag_B(sWp, 1, nt, lane), c);
        c = wmma_f16(a2, frag_B(sWp, 2, nt, lane), c);
        c = wmma_f16(a3, frag_B(sWp, 3, nt, lane), c);
        d[nt] = c;
    }
    __syncthreads();                    // fence before re-using sT as f32
    float* O = (float*)&sT[wave][0];
#pragma unroll
    for (int nt = 0; nt < 4; ++nt) store_D(O, nt * 16, lane, d[nt]);
    __syncthreads();

    if (hi == 0 && (eBase + row) < E) {
        float ss = 0.f;
#pragma unroll
        for (int j0 = 0; j0 < HD; j0 += 4) {
            float4 v = *(float4*)(O + row * OS_ + j0);
            float4 b = *(const float4*)(edge_b + j0);
            v.x += b.x; v.y += b.y; v.z += b.z; v.w += b.w;
            *(float4*)(O + row * OS_ + j0) = v;
            ss = fmaf(v.x, v.x, fmaf(v.y, v.y, fmaf(v.z, v.z, fmaf(v.w, v.w, ss))));
        }
        const float sc = 1.0f / fmaxf(sqrtf(ss), 1e-12f);
        _Float16* dst = x_edge + (size_t)(eBase + row) * HD;
#pragma unroll
        for (int j0 = 0; j0 < HD; j0 += 8) {
            float4 v0 = *(const float4*)(O + row * OS_ + j0);
            float4 v1 = *(const float4*)(O + row * OS_ + j0 + 4);
            v8h p;
            float t0;
            t0 = v0.x * sc; p[0] = (_Float16)((t0 >= 0.f) ? t0 : 0.01f * t0);
            t0 = v0.y * sc; p[1] = (_Float16)((t0 >= 0.f) ? t0 : 0.01f * t0);
            t0 = v0.z * sc; p[2] = (_Float16)((t0 >= 0.f) ? t0 : 0.01f * t0);
            t0 = v0.w * sc; p[3] = (_Float16)((t0 >= 0.f) ? t0 : 0.01f * t0);
            t0 = v1.x * sc; p[4] = (_Float16)((t0 >= 0.f) ? t0 : 0.01f * t0);
            t0 = v1.y * sc; p[5] = (_Float16)((t0 >= 0.f) ? t0 : 0.01f * t0);
            t0 = v1.z * sc; p[6] = (_Float16)((t0 >= 0.f) ? t0 : 0.01f * t0);
            t0 = v1.w * sc; p[7] = (_Float16)((t0 >= 0.f) ? t0 : 0.01f * t0);
            *(v8h*)(dst + j0) = p;
        }
    }
}

// ---------------------------------------------------------------------------
// zero / degree / scatter (segment-sum with native fp32 atomics)
// ---------------------------------------------------------------------------
__global__ void zero_kernel(float* __restrict__ p, long long n) {
    long long i = (long long)blockIdx.x * blockDim.x + threadIdx.x;
    if (i < n) p[i] = 0.f;
}

__global__ void degree_kernel(const int* __restrict__ ei, const int* __restrict__ etype,
                              const int* __restrict__ et3, float* __restrict__ degS,
                              float* __restrict__ degG, int E) {
    int e = blockIdx.x * 256 + threadIdx.x; if (e >= E) return;
    int t = etype[e]; int dst = ei[E + e];
    int e0 = et3[0], e1 = et3[1], e2 = et3[2];
    if (t == e0 || t == e1) unsafeAtomicAdd(&degS[dst], 1.0f);
    if (t == e0 || t == e2) unsafeAtomicAdd(&degG[dst], 1.0f);
}

__global__ __launch_bounds__(256) void scatter_kernel(
    const float* __restrict__ hin, const _Float16* __restrict__ xe,
    const int* __restrict__ ei, const int* __restrict__ etype,
    const int* __restrict__ et3, int sel, float* __restrict__ agg, int E)
{
    long long t = (long long)blockIdx.x * 256 + threadIdx.x;
    int e = (int)(t >> 2); if (e >= E) return;
    int part = ((int)t & 3) * 16;
    int ty = etype[e];
    if (!(ty == et3[0] || ty == et3[sel])) return;
    int src = ei[e], dst = ei[E + e];
    const float*    hs = hin + (size_t)src * HD + part;
    const _Float16* xp = xe  + (size_t)e   * HD + part;
    float*          ap = agg + (size_t)dst * HD + part;
#pragma unroll
    for (int j = 0; j < 16; ++j) unsafeAtomicAdd(ap + j, hs[j] + (float)xp[j]);
}

// ---------------------------------------------------------------------------
// K3: node update: out = agg/max(deg,1) + h @ W(64x64) + b, l2norm, leaky.
// ---------------------------------------------------------------------------
__global__ __launch_bounds__(256) void node_update_kernel(
    const float* __restrict__ hin, const float* __restrict__ agg,
    const float* __restrict__ deg, const _Float16* __restrict__ Wp,
    const float* __restrict__ b64, float* __restrict__ hout, int N)
{
    __shared__ __align__(64) _Float16 sWp[HD * HD];                  // packed, 8 KB
    __shared__ __align__(64) unsigned short sT[8][16 * AS_];
    const int tid = threadIdx.x, wave = tid >> 5, lane = tid & 31;
    stage_weights(Wp, sWp, HD * HD, tid);

    const int row = lane & 15, hi = lane >> 4;
    const int nBase = blockIdx.x * 128 + wave * 16;
    _Float16* A = (_Float16*)&sT[wave][0];
    {
        int n = nBase + row; if (n >= N) n = N - 1;
        const float* hr = hin + (size_t)n * HD + hi * 32;
#pragma unroll
        for (int j0 = 0; j0 < 32; j0 += 8) {
            v8h p; cvt8(p, *(const float4*)(hr + j0), *(const float4*)(hr + j0 + 4));
            *(v8h*)(A + row * AS_ + hi * 32 + j0) = p;
        }
    }
    __syncthreads();

    v16h a0 = frag_A(A, 0, lane), a1 = frag_A(A, 32, lane);
    v8f d[4];
#pragma unroll
    for (int nt = 0; nt < 4; ++nt) {
        v8f c = {};
        c = wmma_f16(a0, frag_B(sWp, 0, nt, lane), c);
        c = wmma_f16(a1, frag_B(sWp, 1, nt, lane), c);
        d[nt] = c;
    }
    __syncthreads();
    float* O = (float*)&sT[wave][0];
#pragma unroll
    for (int nt = 0; nt < 4; ++nt) store_D(O, nt * 16, lane, d[nt]);
    __syncthreads();

    const int n = nBase + row;
    if (hi == 0 && n < N) {
        const float dinv = 1.0f / fmaxf(deg[n], 1.0f);
        const float* ag = agg + (size_t)n * HD;
        float ss = 0.f;
#pragma unroll
        for (int j0 = 0; j0 < HD; j0 += 4) {
            float4 v = *(float4*)(O + row * OS_ + j0);
            float4 b = *(const float4*)(b64 + j0);
            float4 a4 = *(const float4*)(ag + j0);
            v.x += b.x + a4.x * dinv; v.y += b.y + a4.y * dinv;
            v.z += b.z + a4.z * dinv; v.w += b.w + a4.w * dinv;
            *(float4*)(O + row * OS_ + j0) = v;
            ss = fmaf(v.x, v.x, fmaf(v.y, v.y, fmaf(v.z, v.z, fmaf(v.w, v.w, ss))));
        }
        const float sc = 1.0f / fmaxf(sqrtf(ss), 1e-12f);
        float* dst = hout + (size_t)n * HD;
#pragma unroll
        for (int j0 = 0; j0 < HD; j0 += 4) {
            float4 v = *(const float4*)(O + row * OS_ + j0);
            v.x *= sc; v.y *= sc; v.z *= sc; v.w *= sc;
            v.x = (v.x >= 0.f) ? v.x : 0.01f * v.x;
            v.y = (v.y >= 0.f) ? v.y : 0.01f * v.y;
            v.z = (v.z >= 0.f) ? v.z : 0.01f * v.z;
            v.w = (v.w >= 0.f) ? v.w : 0.01f * v.w;
            *(float4*)(dst + j0) = v;
        }
    }
}

// ---------------------------------------------------------------------------
// K4: out = leaky([x_s | x_g](N,128) @ cat_w(128x64) + cat_b)
// ---------------------------------------------------------------------------
__global__ __launch_bounds__(256) void cat_out_kernel(
    const float* __restrict__ xs, const float* __restrict__ xg,
    const _Float16* __restrict__ Wp, const float* __restrict__ bc,
    float* __restrict__ out, int N)
{
    __shared__ __align__(64) _Float16 sWp[128 * HD];
    __shared__ __align__(64) unsigned short sT[8][16 * AS_];
    const int tid = threadIdx.x, wave = tid >> 5, lane = tid & 31;
    stage_weights(Wp, sWp, 128 * HD, tid);

    const int row = lane & 15, hi = lane >> 4;
    const int nBase = blockIdx.x * 128 + wave * 16;
    _Float16* A = (_Float16*)&sT[wave][0];
    {
        int n = nBase + row; if (n >= N) n = N - 1;
        const float* s0 = xs + (size_t)n * HD + hi * 32;
        const float* g0 = xg + (size_t)n * HD + hi * 32;
#pragma unroll
        for (int j0 = 0; j0 < 32; j0 += 8) {
            v8h p; cvt8(p, *(const float4*)(s0 + j0), *(const float4*)(s0 + j0 + 4));
            *(v8h*)(A + row * AS_ + hi * 32 + j0) = p;
        }
#pragma unroll
        for (int j0 = 0; j0 < 32; j0 += 8) {
            v8h p; cvt8(p, *(const float4*)(g0 + j0), *(const float4*)(g0 + j0 + 4));
            *(v8h*)(A + row * AS_ + 64 + hi * 32 + j0) = p;
        }
    }
    __syncthreads();

    v16h a0 = frag_A(A, 0, lane),  a1 = frag_A(A, 32, lane),
         a2 = frag_A(A, 64, lane), a3 = frag_A(A, 96, lane);
    v8f d[4];
#pragma unroll
    for (int nt = 0; nt < 4; ++nt) {
        v8f c = {};
        c = wmma_f16(a0, frag_B(sWp, 0, nt, lane), c);
        c = wmma_f16(a1, frag_B(sWp, 1, nt, lane), c);
        c = wmma_f16(a2, frag_B(sWp, 2, nt, lane), c);
        c = wmma_f16(a3, frag_B(sWp, 3, nt, lane), c);
        d[nt] = c;
    }
    __syncthreads();
    float* O = (float*)&sT[wave][0];
#pragma unroll
    for (int nt = 0; nt < 4; ++nt) store_D(O, nt * 16, lane, d[nt]);
    __syncthreads();

    const int n = nBase + row;
    if (hi == 0 && n < N) {
        float* dst = out + (size_t)n * HD;
#pragma unroll
        for (int j0 = 0; j0 < HD; j0 += 4) {
            float4 v = *(const float4*)(O + row * OS_ + j0);
            float4 b = *(const float4*)(bc + j0);
            v.x += b.x; v.y += b.y; v.z += b.z; v.w += b.w;
            v.x = (v.x >= 0.f) ? v.x : 0.01f * v.x;
            v.y = (v.y >= 0.f) ? v.y : 0.01f * v.y;
            v.z = (v.z >= 0.f) ? v.z : 0.01f * v.z;
            v.w = (v.w >= 0.f) ? v.w : 0.01f * v.w;
            *(float4*)(dst + j0) = v;
        }
    }
}

// ---------------------------------------------------------------------------
extern "C" void kernel_launch(void* const* d_in, const int* in_sizes, int n_in,
                              void* d_out, int out_size, void* d_ws, size_t ws_size,
                              hipStream_t stream) {
    const float* x       = (const float*)d_in[0];
    const int*   ei      = (const int*)d_in[1];
    const int*   et3     = (const int*)d_in[2];
    const int*   etype   = (const int*)d_in[3];
    const float* ew      = (const float*)d_in[4];
    const float* rel_emb = (const float*)d_in[5];
    const float* int_emb = (const float*)d_in[6];
    const float* rbf0_w  = (const float*)d_in[7];
    const float* rbf0_b  = (const float*)d_in[8];
    const float* rbf1_w  = (const float*)d_in[9];
    const float* rbf1_b  = (const float*)d_in[10];
    const float* edge_w  = (const float*)d_in[11];
    const float* edge_b  = (const float*)d_in[12];
    const float* root_w  = (const float*)d_in[13];
    const float* root_b  = (const float*)d_in[14];
    const float* cat_w   = (const float*)d_in[15];
    const float* cat_b   = (const float*)d_in[16];

    const int E = in_sizes[3];
    const int N = in_sizes[0] / HD;

    char* ws = (char*)d_ws;
    auto alloc = [&](size_t bytes) { char* p = ws; ws += (bytes + 255) & ~(size_t)255; return p; };
    _Float16* x_edge  = (_Float16*)alloc((size_t)E * HD * sizeof(_Float16));
    float*    degS    = (float*)alloc((size_t)N * sizeof(float));
    float*    degG    = (float*)alloc((size_t)N * sizeof(float));
    float*    agg     = (float*)alloc((size_t)N * HD * sizeof(float));
    float*    hs1     = (float*)alloc((size_t)N * HD * sizeof(float));
    float*    hs2     = (float*)alloc((size_t)N * HD * sizeof(float));
    float*    hg1     = (float*)alloc((size_t)N * HD * sizeof(float));
    float*    hg2     = (float*)alloc((size_t)N * HD * sizeof(float));
    _Float16* edge_wp = (_Float16*)alloc((size_t)128 * HD * sizeof(_Float16));
    _Float16* root_wp = (_Float16*)alloc((size_t)4 * HD * HD * sizeof(_Float16));
    _Float16* cat_wp  = (_Float16*)alloc((size_t)128 * HD * sizeof(_Float16));

    const dim3 blk(256);
    const long long nh = (long long)N * HD;
    const int gridNode = (N + 127) / 128;
    const int gridZero = (int)((nh + 255) / 256);
    const int gridScat = (int)(((long long)E * 4 + 255) / 256);

    // 0) pack all weight matrices into WMMA-B fragment order (f16)
    pack_w_kernel<<<32, blk, 0, stream>>>(edge_w, edge_wp, 128);
    for (int l = 0; l < 4; ++l)
        pack_w_kernel<<<16, blk, 0, stream>>>(root_w + l * HD * HD, root_wp + l * HD * HD, HD);
    pack_w_kernel<<<32, blk, 0, stream>>>(cat_w, cat_wp, 128);

    // 1) edge embeddings (WMMA GEMM, f16 output)
    edge_embed_kernel<<<(E + 127) / 128, blk, 0, stream>>>(
        et3, etype, ew, rel_emb, int_emb, rbf0_w, rbf0_b, rbf1_w, rbf1_b,
        edge_wp, edge_b, x_edge, E);

    // 2) per-mask in-degrees (constant across layers)
    zero_kernel<<<(N + 255) / 256, blk, 0, stream>>>(degS, N);
    zero_kernel<<<(N + 255) / 256, blk, 0, stream>>>(degG, N);
    degree_kernel<<<(E + 255) / 256, blk, 0, stream>>>(ei, etype, et3, degS, degG, E);

    // 3) s-stack
    zero_kernel<<<gridZero, blk, 0, stream>>>(agg, nh);
    scatter_kernel<<<gridScat, blk, 0, stream>>>(x,   x_edge, ei, etype, et3, 1, agg, E);
    node_update_kernel<<<gridNode, blk, 0, stream>>>(x,   agg, degS, root_wp + 0 * HD * HD, root_b + 0 * HD, hs1, N);
    zero_kernel<<<gridZero, blk, 0, stream>>>(agg, nh);
    scatter_kernel<<<gridScat, blk, 0, stream>>>(hs1, x_edge, ei, etype, et3, 1, agg, E);
    node_update_kernel<<<gridNode, blk, 0, stream>>>(hs1, agg, degS, root_wp + 1 * HD * HD, root_b + 1 * HD, hs2, N);

    // 4) g-stack
    zero_kernel<<<gridZero, blk, 0, stream>>>(agg, nh);
    scatter_kernel<<<gridScat, blk, 0, stream>>>(x,   x_edge, ei, etype, et3, 2, agg, E);
    node_update_kernel<<<gridNode, blk, 0, stream>>>(x,   agg, degG, root_wp + 2 * HD * HD, root_b + 2 * HD, hg1, N);
    zero_kernel<<<gridZero, blk, 0, stream>>>(agg, nh);
    scatter_kernel<<<gridScat, blk, 0, stream>>>(hg1, x_edge, ei, etype, et3, 2, agg, E);
    node_update_kernel<<<gridNode, blk, 0, stream>>>(hg1, agg, degG, root_wp + 3 * HD * HD, root_b + 3 * HD, hg2, N);

    // 5) concat projection -> output
    cat_out_kernel<<<gridNode, blk, 0, stream>>>(hs2, hg2, cat_wp, cat_b, (float*)d_out, N);
}
